// SparseGraphLinear_29798483100096
// MI455X (gfx1250) — matrix operand order repaired and checked by reading, not compiled, not run on hardware
//
#include <hip/hip_runtime.h>

#define TOKENS 8192
#define KDIM   4096   // IN_FEATURES
#define NDIM   4096   // OUT_FEATURES

#define BM 128
#define BN 64
#define BK 32
#define PITCH 40      // bf16 elements per LDS row (32 data + 8 pad) -> conflict-free b128 frag loads

typedef __attribute__((ext_vector_type(16))) __bf16     v16bf;
typedef __attribute__((ext_vector_type(8)))  float      v8f;
typedef __attribute__((ext_vector_type(4)))  unsigned   v4u;
typedef __attribute__((ext_vector_type(8)))  int        v8i;
typedef __attribute__((ext_vector_type(4)))  int        v4i;

__device__ __forceinline__ unsigned short bf16_rne(float f) {
    unsigned u = __float_as_uint(f);
    u += 0x7FFFu + ((u >> 16) & 1u);
    return (unsigned short)(u >> 16);
}

__device__ __forceinline__ void split_bf16(float f, unsigned short& hi, unsigned short& lo) {
    hi = bf16_rne(f);
    float fh = __uint_as_float(((unsigned)hi) << 16);
    lo = bf16_rne(f - fh);
}

// A fragment (16x32 bf16): lane L -> row M = L&15; VGPR0-3 hold K = (L>>4)*8 .. +7,
// VGPR4-7 hold K = 16 + (L>>4)*8 .. +7  (ISA 7.12.2, 16-bit A 16x32)
__device__ __forceinline__ v16bf fragA(const unsigned short* s, int mrow, int lane) {
    const int r = lane & 15, half = lane >> 4;
    const unsigned short* p = s + (mrow + r) * PITCH + half * 8;
    union { v16bf v; v4u u[2]; } f;
    f.u[0] = *(const v4u*)(p);
    f.u[1] = *(const v4u*)(p + 16);
    return f.v;
}

// B fragment (32x16 bf16): lane L -> col N = L&15; VGPR0-3 hold K = (L>>4)*8 .. +7,
// VGPR4-7 hold K = 16 + (L>>4)*8 .. +7, 2 packed per VGPR
__device__ __forceinline__ v16bf fragB(const unsigned short* s, int nrow, int lane) {
    const int c = lane & 15, half = lane >> 4;
    const unsigned short* p = s + (nrow + c) * PITCH + half * 8;
    union { v16bf v; v4u u[2]; } f;
    f.u[0] = *(const v4u*)(p);
    f.u[1] = *(const v4u*)(p + 16);
    return f.v;
}

// ---------------- CDNA5 Tensor Data Mover: 2D tile global -> LDS (TENSORcnt) --------
// D# per CDNA5 ISA. Row: tile_dim0=32 elems (64B) + 16B LDS pad
// (pad_interval=3 -> every 16 DWORDs, pad_amount=3 -> 4 DWORDs) == PITCH 80B.
__device__ __forceinline__ void tdm_load_2d(unsigned long long gaddr, unsigned lds_addr,
                                            unsigned tensor_d0, unsigned tensor_d1,
                                            unsigned tile_d0,   unsigned tile_d1,
                                            unsigned stride_d0)
{
    v4u g0;
    g0[0] = 1u;                                              // count=1 (valid), user mode
    g0[1] = lds_addr;                                        // LDS byte address
    g0[2] = (unsigned)(gaddr & 0xFFFFFFFFull);               // global_addr[31:0]
    g0[3] = (unsigned)((gaddr >> 32) & 0x01FFFFFFull)        // global_addr[56:32]
          | (2u << 30);                                      // type=2 ("image")

    v8i g1;
    g1[0] = (int)((1u << 16)       // data_size = 2 bytes
                | (1u << 20)       // pad_enable
                | (3u << 22)       // pad_interval: 16 DWORDs
                | (3u << 25));     // pad_amount: 4 DWORDs
    g1[1] = (int)((tensor_d0 & 0xFFFFu) << 16);                                  // dim0[15:0]
    g1[2] = (int)(((tensor_d0 >> 16) & 0xFFFFu) | ((tensor_d1 & 0xFFFFu) << 16));// dim0[31:16], dim1[15:0]
    g1[3] = (int)(((tensor_d1 >> 16) & 0xFFFFu) | ((tile_d0 & 0xFFFFu) << 16));  // dim1[31:16], tile_dim0
    g1[4] = (int)(tile_d1 & 0xFFFFu);                                            // tile_dim1 (tile_dim2=0)
    g1[5] = (int)stride_d0;                                                      // dim0_stride[31:0]
    g1[6] = 0;                                                                   // stride hi / dim1_stride
    g1[7] = 0;

    v4i gz = (v4i){0, 0, 0, 0};
#if defined(__clang_major__) && (__clang_major__ >= 23)
    v8i gz8 = (v8i){0, 0, 0, 0, 0, 0, 0, 0};
    __builtin_amdgcn_tensor_load_to_lds(g0, g1, gz, gz, gz8, 0);
#else
    __builtin_amdgcn_tensor_load_to_lds(g0, g1, gz, gz, 0);
#endif
}

// =====================================================================================
// Pre-pass: f32 -> (hi, lo) bf16 split, mask fused on the W side. One-time cost.
// =====================================================================================
__global__ __launch_bounds__(256)
void cvt_x_kernel(const float* __restrict__ X,
                  unsigned short* __restrict__ Xh, unsigned short* __restrict__ Xl) {
    const size_t i4 = ((size_t)blockIdx.x * 256 + threadIdx.x) * 4;
    const float4 v = *(const float4*)(X + i4);
    unsigned short h0, h1, h2, h3, l0, l1, l2, l3;
    split_bf16(v.x, h0, l0); split_bf16(v.y, h1, l1);
    split_bf16(v.z, h2, l2); split_bf16(v.w, h3, l3);
    uint2 hp, lp;
    hp.x = (unsigned)h0 | ((unsigned)h1 << 16); hp.y = (unsigned)h2 | ((unsigned)h3 << 16);
    lp.x = (unsigned)l0 | ((unsigned)l1 << 16); lp.y = (unsigned)l2 | ((unsigned)l3 << 16);
    *(uint2*)(Xh + i4) = hp;
    *(uint2*)(Xl + i4) = lp;
}

__global__ __launch_bounds__(256)
void cvt_w_kernel(const float* __restrict__ W, const int* __restrict__ Msk,
                  unsigned short* __restrict__ Wh, unsigned short* __restrict__ Wl) {
    const size_t i4 = ((size_t)blockIdx.x * 256 + threadIdx.x) * 4;
    const float4 v = *(const float4*)(W + i4);
    const int4  m = *(const int4*)(Msk + i4);
    unsigned short h0, h1, h2, h3, l0, l1, l2, l3;
    split_bf16(v.x * (float)m.x, h0, l0);
    split_bf16(v.y * (float)m.y, h1, l1);
    split_bf16(v.z * (float)m.z, h2, l2);
    split_bf16(v.w * (float)m.w, h3, l3);
    uint2 hp, lp;
    hp.x = (unsigned)h0 | ((unsigned)h1 << 16); hp.y = (unsigned)h2 | ((unsigned)h3 << 16);
    lp.x = (unsigned)l0 | ((unsigned)l1 << 16); lp.y = (unsigned)l2 | ((unsigned)l3 << 16);
    *(uint2*)(Wh + i4) = hp;
    *(uint2*)(Wl + i4) = lp;
}

// =====================================================================================
// Main GEMM on pre-split bf16: TDM global->LDS double buffering + bf16x3 WMMA.
// Waves 0..3 each drive one tile stream (Ah/Al/Bh/Bl) on their own TENSORcnt.
// =====================================================================================
__global__ __launch_bounds__(256)
void gemm_bf16x3_kernel(const unsigned short* __restrict__ Xh,
                        const unsigned short* __restrict__ Xl,
                        const unsigned short* __restrict__ Wh,
                        const unsigned short* __restrict__ Wl,
                        const float* __restrict__ Bias,
                        float* __restrict__ Out)
{
    __shared__ unsigned short sAh[2][BM * PITCH];
    __shared__ unsigned short sAl[2][BM * PITCH];
    __shared__ unsigned short sBh[2][BN * PITCH];
    __shared__ unsigned short sBl[2][BN * PITCH];

    const int t     = threadIdx.x;
    const int lane  = t & 31;
    const int wave  = t >> 5;
    const int waveM = wave & 3;
    const int waveN = wave >> 2;
    const int mBase = blockIdx.y * BM;
    const int nBase = blockIdx.x * BN;

    // Low 32 bits of the flat shared address == LDS byte offset (aperture is in [63:32])
    const unsigned ldsAh0 = (unsigned)(uintptr_t)(&sAh[0][0]);
    const unsigned ldsAl0 = (unsigned)(uintptr_t)(&sAl[0][0]);
    const unsigned ldsBh0 = (unsigned)(uintptr_t)(&sBh[0][0]);
    const unsigned ldsBl0 = (unsigned)(uintptr_t)(&sBl[0][0]);

    // Per-wave TDM stream setup (waves 0..3), all values wave-uniform -> SGPRs.
    const bool issuer = (wave < 4);
    unsigned long long gtile0 = 0;   // VA of this stream's tile column at k=0
    unsigned ldsbase = 0, bufSz = 0, tileRows = 0, tensRows = 0;
    if (issuer) {
        if (wave == 0) {
            gtile0 = (unsigned long long)(uintptr_t)Xh + (unsigned long long)mBase * KDIM * 2ull;
            ldsbase = ldsAh0; bufSz = (unsigned)(BM * PITCH) * 2u; tileRows = BM; tensRows = TOKENS;
        } else if (wave == 1) {
            gtile0 = (unsigned long long)(uintptr_t)Xl + (unsigned long long)mBase * KDIM * 2ull;
            ldsbase = ldsAl0; bufSz = (unsigned)(BM * PITCH) * 2u; tileRows = BM; tensRows = TOKENS;
        } else if (wave == 2) {
            gtile0 = (unsigned long long)(uintptr_t)Wh + (unsigned long long)nBase * KDIM * 2ull;
            ldsbase = ldsBh0; bufSz = (unsigned)(BN * PITCH) * 2u; tileRows = BN; tensRows = NDIM;
        } else {
            gtile0 = (unsigned long long)(uintptr_t)Wl + (unsigned long long)nBase * KDIM * 2ull;
            ldsbase = ldsBl0; bufSz = (unsigned)(BN * PITCH) * 2u; tileRows = BN; tensRows = NDIM;
        }
    }

    v8f acc[2][2];
#pragma unroll
    for (int mi = 0; mi < 2; ++mi)
#pragma unroll
        for (int ni = 0; ni < 2; ++ni)
            acc[mi][ni] = (v8f){0.f, 0.f, 0.f, 0.f, 0.f, 0.f, 0.f, 0.f};

    if (issuer) {
        tdm_load_2d(gtile0, ldsbase, KDIM, tensRows, BK, tileRows, KDIM);
        __builtin_amdgcn_s_wait_tensorcnt(0);
    }
    __syncthreads();

    const int NSTEP = KDIM / BK;  // 128
    for (int kt = 0; kt < NSTEP; ++kt) {
        const int cur = kt & 1;
        if (kt + 1 < NSTEP && issuer) {
            const unsigned nb = (unsigned)(cur ^ 1);
            tdm_load_2d(gtile0 + (unsigned long long)((kt + 1) * BK) * 2ull,
                        ldsbase + nb * bufSz, KDIM, tensRows, BK, tileRows, KDIM);
        }

        v16bf ah[2], al[2], bh[2], bl[2];
#pragma unroll
        for (int mi = 0; mi < 2; ++mi) {
            ah[mi] = fragA(sAh[cur], waveM * 32 + mi * 16, lane);
            al[mi] = fragA(sAl[cur], waveM * 32 + mi * 16, lane);
        }
#pragma unroll
        for (int ni = 0; ni < 2; ++ni) {
            bh[ni] = fragB(sBh[cur], waveN * 32 + ni * 16, lane);
            bl[ni] = fragB(sBl[cur], waveN * 32 + ni * 16, lane);
        }

#pragma unroll
        for (int mi = 0; mi < 2; ++mi)
#pragma unroll
            for (int ni = 0; ni < 2; ++ni) {
                acc[mi][ni] = __builtin_amdgcn_wmma_f32_16x16x32_bf16(
                    false, ah[mi], false, bh[ni], (short)0, acc[mi][ni], false, false);
                acc[mi][ni] = __builtin_amdgcn_wmma_f32_16x16x32_bf16(
                    false, ah[mi], false, bl[ni], (short)0, acc[mi][ni], false, false);
                acc[mi][ni] = __builtin_amdgcn_wmma_f32_16x16x32_bf16(
                    false, al[mi], false, bh[ni], (short)0, acc[mi][ni], false, false);
            }

        if (issuer) __builtin_amdgcn_s_wait_tensorcnt(0);
        __syncthreads();
    }

    const int colBase = nBase + waveN * 32 + (lane & 15);
    const int rowBase = mBase + waveM * 32 + (lane >> 4) * 8;
#pragma unroll
    for (int mi = 0; mi < 2; ++mi)
#pragma unroll
        for (int ni = 0; ni < 2; ++ni) {
            const int n  = colBase + ni * 16;
            const int m0 = rowBase + mi * 16;
            const float bv = Bias[n];
#pragma unroll
            for (int r = 0; r < 8; ++r)
                Out[(size_t)(m0 + r) * NDIM + n] = acc[mi][ni][r] + bv;
        }
}

// =====================================================================================
// Fallback: fused mask+split+GEMM, used if workspace is too small.
// =====================================================================================
__device__ __forceinline__ void stage_load_fused(
    const float* __restrict__ X, const float* __restrict__ W, const int* __restrict__ Msk,
    unsigned short* __restrict__ sAh, unsigned short* __restrict__ sAl,
    unsigned short* __restrict__ sBh, unsigned short* __restrict__ sBl,
    int mBase, int nBase, int k0, int t)
{
    const int c4 = (t & 7) * 4;
    const int r0 = t >> 3;
#pragma unroll
    for (int p = 0; p < 4; ++p) {
        const int row = r0 + p * 32;
        const float4 v = *(const float4*)(X + (size_t)(mBase + row) * KDIM + k0 + c4);
        unsigned short h0, h1, h2, h3, l0, l1, l2, l3;
        split_bf16(v.x, h0, l0); split_bf16(v.y, h1, l1);
        split_bf16(v.z, h2, l2); split_bf16(v.w, h3, l3);
        uint2 hp, lp;
        hp.x = (unsigned)h0 | ((unsigned)h1 << 16); hp.y = (unsigned)h2 | ((unsigned)h3 << 16);
        lp.x = (unsigned)l0 | ((unsigned)l1 << 16); lp.y = (unsigned)l2 | ((unsigned)l3 << 16);
        *(uint2*)(sAh + row * PITCH + c4) = hp;
        *(uint2*)(sAl + row * PITCH + c4) = lp;
    }
#pragma unroll
    for (int p = 0; p < 2; ++p) {
        const int row = r0 + p * 32;
        const size_t off = (size_t)(nBase + row) * KDIM + k0 + c4;
        const float4 wv = *(const float4*)(W + off);
        const int4  mv = *(const int4*)(Msk + off);
        unsigned short h0, h1, h2, h3, l0, l1, l2, l3;
        split_bf16(wv.x * (float)mv.x, h0, l0);
        split_bf16(wv.y * (float)mv.y, h1, l1);
        split_bf16(wv.z * (float)mv.z, h2, l2);
        split_bf16(wv.w * (float)mv.w, h3, l3);
        uint2 hp, lp;
        hp.x = (unsigned)h0 | ((unsigned)h1 << 16); hp.y = (unsigned)h2 | ((unsigned)h3 << 16);
        lp.x = (unsigned)l0 | ((unsigned)l1 << 16); lp.y = (unsigned)l2 | ((unsigned)l3 << 16);
        *(uint2*)(sBh + row * PITCH + c4) = hp;
        *(uint2*)(sBl + row * PITCH + c4) = lp;
    }
}

__global__ __launch_bounds__(256)
void gemm_fused_kernel(const float* __restrict__ X, const float* __restrict__ W,
                       const float* __restrict__ Bias, const int* __restrict__ Msk,
                       float* __restrict__ Out)
{
    __shared__ unsigned short sAh[2][BM * PITCH];
    __shared__ unsigned short sAl[2][BM * PITCH];
    __shared__ unsigned short sBh[2][BN * PITCH];
    __shared__ unsigned short sBl[2][BN * PITCH];

    const int t     = threadIdx.x;
    const int lane  = t & 31;
    const int wave  = t >> 5;
    const int waveM = wave & 3;
    const int waveN = wave >> 2;
    const int mBase = blockIdx.y * BM;
    const int nBase = blockIdx.x * BN;

    v8f acc[2][2];
#pragma unroll
    for (int mi = 0; mi < 2; ++mi)
#pragma unroll
        for (int ni = 0; ni < 2; ++ni)
            acc[mi][ni] = (v8f){0.f, 0.f, 0.f, 0.f, 0.f, 0.f, 0.f, 0.f};

    stage_load_fused(X, W, Msk, sAh[0], sAl[0], sBh[0], sBl[0], mBase, nBase, 0, t);
    __syncthreads();

    const int NSTEP = KDIM / BK;
    for (int kt = 0; kt < NSTEP; ++kt) {
        const int cur = kt & 1;
        if (kt + 1 < NSTEP) {
            const int nxt = cur ^ 1;
            stage_load_fused(X, W, Msk, sAh[nxt], sAl[nxt], sBh[nxt], sBl[nxt],
                             mBase, nBase, (kt + 1) * BK, t);
        }
        v16bf ah[2], al[2], bh[2], bl[2];
#pragma unroll
        for (int mi = 0; mi < 2; ++mi) {
            ah[mi] = fragA(sAh[cur], waveM * 32 + mi * 16, lane);
            al[mi] = fragA(sAl[cur], waveM * 32 + mi * 16, lane);
        }
#pragma unroll
        for (int ni = 0; ni < 2; ++ni) {
            bh[ni] = fragB(sBh[cur], waveN * 32 + ni * 16, lane);
            bl[ni] = fragB(sBl[cur], waveN * 32 + ni * 16, lane);
        }
#pragma unroll
        for (int mi = 0; mi < 2; ++mi)
#pragma unroll
            for (int ni = 0; ni < 2; ++ni) {
                acc[mi][ni] = __builtin_amdgcn_wmma_f32_16x16x32_bf16(
                    false, ah[mi], false, bh[ni], (short)0, acc[mi][ni], false, false);
                acc[mi][ni] = __builtin_amdgcn_wmma_f32_16x16x32_bf16(
                    false, ah[mi], false, bl[ni], (short)0, acc[mi][ni], false, false);
                acc[mi][ni] = __builtin_amdgcn_wmma_f32_16x16x32_bf16(
                    false, al[mi], false, bh[ni], (short)0, acc[mi][ni], false, false);
            }
        __syncthreads();
    }

    const int colBase = nBase + waveN * 32 + (lane & 15);
    const int rowBase = mBase + waveM * 32 + (lane >> 4) * 8;
#pragma unroll
    for (int mi = 0; mi < 2; ++mi)
#pragma unroll
        for (int ni = 0; ni < 2; ++ni) {
            const int n  = colBase + ni * 16;
            const int m0 = rowBase + mi * 16;
            const float bv = Bias[n];
#pragma unroll
            for (int r = 0; r < 8; ++r)
                Out[(size_t)(m0 + r) * NDIM + n] = acc[mi][ni][r] + bv;
        }
}

extern "C" void kernel_launch(void* const* d_in, const int* in_sizes, int n_in,
                              void* d_out, int out_size, void* d_ws, size_t ws_size,
                              hipStream_t stream) {
    (void)in_sizes; (void)n_in; (void)out_size;
    const float* X    = (const float*)d_in[0];
    const float* W    = (const float*)d_in[1];
    const float* Bias = (const float*)d_in[2];
    const int*   Msk  = (const int*)d_in[3];
    float* Out = (float*)d_out;

    const size_t nX = (size_t)TOKENS * KDIM;
    const size_t nW = (size_t)NDIM * KDIM;
    const size_t WS_NEED = (2 * nX + 2 * nW) * sizeof(unsigned short);  // 192 MB

    dim3 grid(NDIM / BN, TOKENS / BM);  // (64, 64)

    if (ws_size >= WS_NEED) {
        unsigned short* Xh = (unsigned short*)d_ws;
        unsigned short* Xl = Xh + nX;
        unsigned short* Wh = Xl + nX;
        unsigned short* Wl = Wh + nW;
        cvt_x_kernel<<<(unsigned)(nX / 1024), 256, 0, stream>>>(X, Xh, Xl);
        cvt_w_kernel<<<(unsigned)(nW / 1024), 256, 0, stream>>>(W, Msk, Wh, Wl);
        gemm_bf16x3_kernel<<<grid, 256, 0, stream>>>(Xh, Xl, Wh, Wl, Bias, Out);
    } else {
        gemm_fused_kernel<<<grid, 256, 0, stream>>>(X, W, Bias, Msk, Out);
    }
}
